// GNS_68272800137598
// MI455X (gfx1250) — matrix operand order repaired and checked by reading, not compiled, not run on hardware
//
#include <hip/hip_runtime.h>
#include <hip/hip_bf16.h>
#include <cstdint>

// ---------------------------------------------------------------------------
// GNS (GraphNetwork simulator) forward pass for gfx1250 (MI455X, wave32).
// All matmuls run on v_wmma_f32_16x16x32_f16 with f32 accumulation; graph
// state (node/edge latents, aggregation) stays f32 for residual + LayerNorm.
// ---------------------------------------------------------------------------

#define LATENT   128
#define TM        64      // rows (edges or nodes) per workgroup
#define NT         8      // 128/16 output column tiles per wave
#define NTHREADS 128      // 4 wave32s; each wave owns 16 rows

typedef __attribute__((ext_vector_type(16))) _Float16 v16h;
typedef __attribute__((ext_vector_type(8)))  _Float16 v8h;
typedef __attribute__((ext_vector_type(8)))  float    v8f;

// ------------------------------- device helpers ----------------------------

__device__ __forceinline__ int lane_id() { return threadIdx.x & 31; }

// A-fragment (16x32 f16) from row-major f16 LDS tile, per CDNA5 ISA layout:
// lanes 0-15: row M=lane, halves 0..7 = K kb+0..7, halves 8..15 = K kb+16..23
// lanes 16-31: row M=lane-16, K ranges shifted by +8.
__device__ __forceinline__ v16h load_a_frag(const _Float16* lds, int row0,
                                            int kbase, int stride)
{
  const int lane = lane_id();
  const _Float16* p = lds + (size_t)(row0 + (lane & 15)) * stride
                          + kbase + ((lane >> 4) << 3);
  v8h lo = *(const v8h*)(p);
  v8h hi = *(const v8h*)(p + 16);
  v16h a;
#pragma unroll
  for (int i = 0; i < 8; ++i) { a[i] = lo[i]; a[i + 8] = hi[i]; }
  return a;
}

// GEMM for one wave: C[16 x NTILES*16] += A[16 x KCH*32] * B, B pre-packed in
// fragment order (frag = kc*NTILES + t; 512 halves per fragment).
template<int KCH, int NTILES>
__device__ __forceinline__ void wave_gemm(v8f acc[NTILES], const _Float16* ldsA,
                                          int row0, int astride,
                                          const _Float16* __restrict__ packB)
{
  const int lane = lane_id();
  for (int kc = 0; kc < KCH; ++kc) {
    v16h a = load_a_frag(ldsA, row0, kc * 32, astride);
#pragma unroll
    for (int t = 0; t < NTILES; ++t) {
      v16h b = *(const v16h*)(packB + (size_t)(kc * NTILES + t) * 512 + lane * 16);
      acc[t] = __builtin_amdgcn_wmma_f32_16x16x32_f16(
          false, a, false, b, (short)0, acc[t], false, false);
    }
  }
}

// D tile (16x16 f32: lane -> col = t*16 + lane%16; vgpr r -> row r (+8 for
// upper half-lanes)) -> bias + ReLU -> f16 LDS row-major (stride LATENT).
__device__ __forceinline__ void store_bias_relu_f16(_Float16* lds, const v8f& c,
                                                    const float* __restrict__ bias,
                                                    int row0, int t)
{
  const int lane = lane_id();
  const int col  = t * 16 + (lane & 15);
  const float bv = bias[col];
  const int r0   = row0 + ((lane >> 4) << 3);
#pragma unroll
  for (int r = 0; r < 8; ++r) {
    float v = c[r] + bv;
    lds[(r0 + r) * LATENT + col] = (_Float16)(v > 0.f ? v : 0.f);
  }
}

__device__ __forceinline__ void store_bias_f32(float* lds, const v8f& c,
                                               const float* __restrict__ bias,
                                               int row0, int t)
{
  const int lane = lane_id();
  const int col  = t * 16 + (lane & 15);
  const float bv = bias[col];
  const int r0   = row0 + ((lane >> 4) << 3);
#pragma unroll
  for (int r = 0; r < 8; ++r)
    lds[(r0 + r) * LATENT + col] = c[r] + bv;
}

// 3-layer MLP (in -> 128 -> 128 -> 128), ReLU between layers, final layer
// (pre-LayerNorm, bias added) written as f32 to outF.
template<int KCH1>
__device__ __forceinline__ void mlp3_to_f32(
    const _Float16* bufIn, int strideIn,
    _Float16* actA, _Float16* actB, float* outF,
    const _Float16* pw1, const float* b1,
    const _Float16* pw2, const float* b2,
    const _Float16* pw3, const float* b3)
{
  const int row0 = (threadIdx.x >> 5) * 16;
  const v8f vzero = {0.f, 0.f, 0.f, 0.f, 0.f, 0.f, 0.f, 0.f};
  v8f acc[NT];

#pragma unroll
  for (int t = 0; t < NT; ++t) acc[t] = vzero;
  wave_gemm<KCH1, NT>(acc, bufIn, row0, strideIn, pw1);
#pragma unroll
  for (int t = 0; t < NT; ++t) store_bias_relu_f16(actA, acc[t], b1, row0, t);
  __syncthreads();

#pragma unroll
  for (int t = 0; t < NT; ++t) acc[t] = vzero;
  wave_gemm<4, NT>(acc, actA, row0, LATENT, pw2);
#pragma unroll
  for (int t = 0; t < NT; ++t) store_bias_relu_f16(actB, acc[t], b2, row0, t);
  __syncthreads();

#pragma unroll
  for (int t = 0; t < NT; ++t) acc[t] = vzero;
  wave_gemm<4, NT>(acc, actB, row0, LATENT, pw3);
#pragma unroll
  for (int t = 0; t < NT; ++t) store_bias_f32(outF, acc[t], b3, row0, t);
  __syncthreads();
}

// Per-row LayerNorm stats over 128 columns (population variance, eps 1e-5).
__device__ __forceinline__ void ln_stats(const float* outF, float* mu, float* rs)
{
  const int tid = threadIdx.x;
  if (tid < TM) {
    const float* row = outF + tid * LATENT;
    float s = 0.f, s2 = 0.f;
    for (int c = 0; c < LATENT; ++c) { float v = row[c]; s += v; s2 += v * v; }
    const float m = s * (1.f / LATENT);
    mu[tid] = m;
    rs[tid] = rsqrtf(s2 * (1.f / LATENT) - m * m + 1e-5f);
  }
  __syncthreads();
}

// ------------------------------- kernels -----------------------------------

// Repack f32 weight [Korig x Norig] (row-major) into f16 WMMA-B fragment
// order, zero-padded to [Kp x Np]. frag f = kc*(Np/16)+t; within a fragment:
// lane 0-31 x halves 0-15; col = t*16 + lane%16; k = kc*32 + (lane/16)*8 + (e%8) + (e/8)*16.
__global__ void k_repack(const float* __restrict__ W, _Float16* __restrict__ dst,
                         int Korig, int Norig, int Kp, int Np)
{
  const int idx = blockIdx.x * blockDim.x + threadIdx.x;
  if (idx >= Kp * Np) return;
  const int e    = idx & 15;
  const int lane = (idx >> 4) & 31;
  const int frag = idx >> 9;
  const int ntiles = Np >> 4;
  const int kc = frag / ntiles;
  const int t  = frag - kc * ntiles;
  const int n  = t * 16 + (lane & 15);
  const int k  = kc * 32 + ((lane >> 4) << 3) + (e & 7) + ((e >> 3) << 4);
  const float v = (k < Korig && n < Norig) ? W[(size_t)k * Norig + n] : 0.f;
  dst[idx] = (_Float16)v;
}

__global__ void k_zero(float* __restrict__ p, long long nElem)
{
  const long long i = (long long)blockIdx.x * blockDim.x + threadIdx.x;
  if (i < nElem) p[i] = 0.f;
}

// Encoder (nodes): concat(node_features[30], embed[ptype][16]) -> MLP -> LN.
__global__ __launch_bounds__(NTHREADS) void k_enc_node(
    const float* __restrict__ feat, const int* __restrict__ ptype,
    const float* __restrict__ embed,
    const _Float16* pw1, const float* b1,
    const _Float16* pw2, const float* b2,
    const _Float16* pw3, const float* b3,
    const float* g, const float* o,
    float* __restrict__ nodesOut, int n)
{
  extern __shared__ char smem[];
  _Float16* bufIn = (_Float16*)smem;            // TM x 64 (K padded 46->64)
  _Float16* actA  = bufIn + TM * 64;
  _Float16* actB  = actA + TM * LATENT;
  float*    outF  = (float*)(actB + TM * LATENT);
  float*    mu    = outF + TM * LATENT;
  float*    rs    = mu + TM;

  const int tid  = threadIdx.x;
  const int base = blockIdx.x * TM;

  for (int idx = tid; idx < TM * 64; idx += NTHREADS) {
    const int row = idx >> 6, col = idx & 63;
    const int nd = base + row;
    float v = 0.f;
    if (nd < n) {
      if (col < 30)      v = feat[(size_t)nd * 30 + col];
      else if (col < 46) v = embed[ptype[nd] * 16 + (col - 30)];
    }
    bufIn[idx] = (_Float16)v;
  }
  __syncthreads();

  mlp3_to_f32<2>(bufIn, 64, actA, actB, outF, pw1, b1, pw2, b2, pw3, b3);
  ln_stats(outF, mu, rs);

  for (int idx = tid; idx < TM * LATENT; idx += NTHREADS) {
    const int row = idx >> 7, col = idx & 127;
    const int nd = base + row;
    if (nd < n)
      nodesOut[(size_t)nd * LATENT + col] =
          (outF[idx] - mu[row]) * rs[row] * g[col] + o[col];
  }
}

// Encoder (edges): edge_features[4] -> MLP -> LN.
__global__ __launch_bounds__(NTHREADS) void k_enc_edge(
    const float* __restrict__ efeat,
    const _Float16* pw1, const float* b1,
    const _Float16* pw2, const float* b2,
    const _Float16* pw3, const float* b3,
    const float* g, const float* o,
    float* __restrict__ edgesOut, int nE)
{
  extern __shared__ char smem[];
  _Float16* bufIn = (_Float16*)smem;            // TM x 32 (K padded 4->32)
  _Float16* actA  = bufIn + TM * 32;
  _Float16* actB  = actA + TM * LATENT;
  float*    outF  = (float*)(actB + TM * LATENT);
  float*    mu    = outF + TM * LATENT;
  float*    rs    = mu + TM;

  const int tid  = threadIdx.x;
  const int base = blockIdx.x * TM;

  for (int idx = tid; idx < TM * 32; idx += NTHREADS) {
    const int row = idx >> 5, col = idx & 31;
    const int e = base + row;
    float v = 0.f;
    if (e < nE && col < 4) v = efeat[(size_t)e * 4 + col];
    bufIn[idx] = (_Float16)v;
  }
  __syncthreads();

  mlp3_to_f32<1>(bufIn, 32, actA, actB, outF, pw1, b1, pw2, b2, pw3, b3);
  ln_stats(outF, mu, rs);

  for (int idx = tid; idx < TM * LATENT; idx += NTHREADS) {
    const int row = idx >> 7, col = idx & 127;
    const int e = base + row;
    if (e < nE)
      edgesOut[(size_t)e * LATENT + col] =
          (outF[idx] - mu[row]) * rs[row] * g[col] + o[col];
  }
}

// Processor edge block: gather [send|recv|edge] (K=384) -> MLP -> LN,
// edge residual update, atomic scatter-add into agg[receiver].
__global__ __launch_bounds__(NTHREADS) void k_edge_step(
    const float* __restrict__ nodes, float* __restrict__ edges,
    float* __restrict__ agg,
    const int* __restrict__ senders, const int* __restrict__ receivers,
    const _Float16* pw1, const float* b1,
    const _Float16* pw2, const float* b2,
    const _Float16* pw3, const float* b3,
    const float* g, const float* o, int nE)
{
  extern __shared__ char smem[];
  _Float16* bufIn = (_Float16*)smem;            // TM x 384
  _Float16* actA  = bufIn + TM * 384;
  _Float16* actB  = actA + TM * LATENT;
  float*    outF  = (float*)(actB + TM * LATENT);
  float*    mu    = outF + TM * LATENT;
  float*    rs    = mu + TM;
  int*      sIdx  = (int*)(rs + TM);
  int*      rIdx  = sIdx + TM;

  const int tid  = threadIdx.x;
  const int base = blockIdx.x * TM;

  if (tid < TM) {
    const int e = base + tid;
    sIdx[tid] = (e < nE) ? senders[e]   : 0;
    rIdx[tid] = (e < nE) ? receivers[e] : 0;
  }
  __syncthreads();

  for (int idx = tid; idx < TM * LATENT; idx += NTHREADS) {
    const int row = idx >> 7, col = idx & 127;
    const int e = base + row;
    const bool valid = e < nE;
    _Float16* dst = bufIn + (size_t)row * 384;
    dst[col]       = (_Float16)(valid ? nodes[(size_t)sIdx[row] * LATENT + col] : 0.f);
    dst[128 + col] = (_Float16)(valid ? nodes[(size_t)rIdx[row] * LATENT + col] : 0.f);
    dst[256 + col] = (_Float16)(valid ? edges[(size_t)e * LATENT + col] : 0.f);
  }
  __syncthreads();

  mlp3_to_f32<12>(bufIn, 384, actA, actB, outF, pw1, b1, pw2, b2, pw3, b3);
  ln_stats(outF, mu, rs);

  for (int idx = tid; idx < TM * LATENT; idx += NTHREADS) {
    const int row = idx >> 7, col = idx & 127;
    const int e = base + row;
    if (e < nE) {
      const float eu = (outF[idx] - mu[row]) * rs[row] * g[col] + o[col];
      const size_t ei = (size_t)e * LATENT + col;
      edges[ei] += eu;                                       // edge residual
      atomicAdd(&agg[(size_t)rIdx[row] * LATENT + col], eu); // segment_sum
    }
  }
}

// Processor node block: concat(nodes, agg) (K=256) -> MLP -> LN, residual.
__global__ __launch_bounds__(NTHREADS) void k_node_step(
    float* __restrict__ nodes, const float* __restrict__ agg,
    const _Float16* pw1, const float* b1,
    const _Float16* pw2, const float* b2,
    const _Float16* pw3, const float* b3,
    const float* g, const float* o, int n)
{
  extern __shared__ char smem[];
  _Float16* bufIn = (_Float16*)smem;            // TM x 256
  _Float16* actA  = bufIn + TM * 256;
  _Float16* actB  = actA + TM * LATENT;
  float*    outF  = (float*)(actB + TM * LATENT);
  float*    mu    = outF + TM * LATENT;
  float*    rs    = mu + TM;

  const int tid  = threadIdx.x;
  const int base = blockIdx.x * TM;

  for (int idx = tid; idx < TM * LATENT; idx += NTHREADS) {
    const int row = idx >> 7, col = idx & 127;
    const int nd = base + row;
    const bool valid = nd < n;
    _Float16* dst = bufIn + (size_t)row * 256;
    dst[col]       = (_Float16)(valid ? nodes[(size_t)nd * LATENT + col] : 0.f);
    dst[128 + col] = (_Float16)(valid ? agg[(size_t)nd * LATENT + col] : 0.f);
  }
  __syncthreads();

  mlp3_to_f32<8>(bufIn, 256, actA, actB, outF, pw1, b1, pw2, b2, pw3, b3);
  ln_stats(outF, mu, rs);

  for (int idx = tid; idx < TM * LATENT; idx += NTHREADS) {
    const int row = idx >> 7, col = idx & 127;
    const int nd = base + row;
    if (nd < n) {
      const float nu = (outF[idx] - mu[row]) * rs[row] * g[col] + o[col];
      nodes[(size_t)nd * LATENT + col] += nu;                // node residual
    }
  }
}

// Decoder: 128 -> 128 -> 128 -> 3 (no LayerNorm). Final N padded 3->16.
__global__ __launch_bounds__(NTHREADS) void k_decode(
    const float* __restrict__ nodes,
    const _Float16* pw1, const float* b1,
    const _Float16* pw2, const float* b2,
    const _Float16* pw3, const float* b3,
    float* __restrict__ out, int n)
{
  extern __shared__ char smem[];
  _Float16* bufIn = (_Float16*)smem;            // TM x 128
  _Float16* actA  = bufIn + TM * LATENT;
  _Float16* actB  = actA + TM * LATENT;

  const int tid  = threadIdx.x;
  const int base = blockIdx.x * TM;

  for (int idx = tid; idx < TM * LATENT; idx += NTHREADS) {
    const int row = idx >> 7, col = idx & 127;
    const int nd = base + row;
    bufIn[idx] = (_Float16)((nd < n) ? nodes[(size_t)nd * LATENT + col] : 0.f);
  }
  __syncthreads();

  const int lane = lane_id();
  const int row0 = (tid >> 5) * 16;
  const v8f vzero = {0.f, 0.f, 0.f, 0.f, 0.f, 0.f, 0.f, 0.f};
  v8f acc[NT];

#pragma unroll
  for (int t = 0; t < NT; ++t) acc[t] = vzero;
  wave_gemm<4, NT>(acc, bufIn, row0, LATENT, pw1);
#pragma unroll
  for (int t = 0; t < NT; ++t) store_bias_relu_f16(actA, acc[t], b1, row0, t);
  __syncthreads();

#pragma unroll
  for (int t = 0; t < NT; ++t) acc[t] = vzero;
  wave_gemm<4, NT>(acc, actA, row0, LATENT, pw2);
#pragma unroll
  for (int t = 0; t < NT; ++t) store_bias_relu_f16(actB, acc[t], b2, row0, t);
  __syncthreads();

  v8f acc3[1] = {vzero};
  wave_gemm<4, 1>(acc3, actB, row0, LATENT, pw3);

  const int col = lane & 15;
  if (col < 3) {
    const int r0 = row0 + ((lane >> 4) << 3);
#pragma unroll
    for (int r = 0; r < 8; ++r) {
      const int nd = base + r0 + r;
      if (nd < n) out[(size_t)nd * 3 + col] = acc3[0][r] + b3[col];
    }
  }
}

// ------------------------------- host ---------------------------------------

extern "C" void kernel_launch(void* const* d_in, const int* in_sizes, int n_in,
                              void* d_out, int out_size, void* d_ws, size_t ws_size,
                              hipStream_t stream)
{
  // Tail inputs (dict order): node_features, edge_features, senders,
  // receivers, particle_type. Params are the leading n_in-5 leaves in
  // insertion order: embed, enc_node{W0..2,b0..2,g,o}, enc_edge{...},
  // proc_edge[0..9]{...}, proc_node[0..9]{...}, dec{W0..2,b0..2}.
  const int NB = n_in - 5;
  const float* node_feat = (const float*)d_in[NB + 0];
  const float* edge_feat = (const float*)d_in[NB + 1];
  const int*   senders   = (const int*)  d_in[NB + 2];
  const int*   receivers = (const int*)  d_in[NB + 3];
  const int*   ptype     = (const int*)  d_in[NB + 4];

  const int nNodes = in_sizes[NB + 0] / 30;
  const int nEdges = in_sizes[NB + 1] / 4;

  auto F = [&](int i) { return (const float*)d_in[i]; };

  const int I_EMB       = 0;
  const int I_ENC_NODE  = 1;    // W0,W1,W2,b0,b1,b2,g,o
  const int I_ENC_EDGE  = 9;
  const int I_PROC_EDGE = 17;   // 10 x 8 leaves
  const int I_PROC_NODE = 97;   // 10 x 8 leaves
  const int I_DEC       = 177;  // W0,W1,W2,b0,b1,b2

  // ---- workspace carve ----
  char* w = (char*)d_ws;
  float* nodes = (float*)w; w += (size_t)nNodes * LATENT * sizeof(float);
  float* edges = (float*)w; w += (size_t)nEdges * LATENT * sizeof(float);
  float* agg   = (float*)w; w += (size_t)nNodes * LATENT * sizeof(float);
  w = (char*)(((uintptr_t)w + 1023) & ~(uintptr_t)1023);
  _Float16* packBase = (_Float16*)w;

  size_t packOff = 0;
  auto repack = [&](int widx, int K, int N, int Kp, int Np) -> const _Float16* {
    _Float16* dst = packBase + packOff;
    const int total = Kp * Np;
    k_repack<<<dim3((total + 255) / 256), dim3(256), 0, stream>>>(
        F(widx), dst, K, N, Kp, Np);
    packOff += (size_t)total;
    return dst;
  };

  // ---- weight repack (f32 -> f16 WMMA-B fragment order) ----
  const _Float16* en_w1 = repack(I_ENC_NODE + 0, 46, 128, 64, 128);
  const _Float16* en_w2 = repack(I_ENC_NODE + 1, 128, 128, 128, 128);
  const _Float16* en_w3 = repack(I_ENC_NODE + 2, 128, 128, 128, 128);
  const _Float16* ee_w1 = repack(I_ENC_EDGE + 0, 4, 128, 32, 128);
  const _Float16* ee_w2 = repack(I_ENC_EDGE + 1, 128, 128, 128, 128);
  const _Float16* ee_w3 = repack(I_ENC_EDGE + 2, 128, 128, 128, 128);
  const _Float16 *pe1[10], *pe2[10], *pe3[10], *pn1[10], *pn2[10], *pn3[10];
  for (int i = 0; i < 10; ++i) {
    const int be = I_PROC_EDGE + 8 * i;
    pe1[i] = repack(be + 0, 384, 128, 384, 128);
    pe2[i] = repack(be + 1, 128, 128, 128, 128);
    pe3[i] = repack(be + 2, 128, 128, 128, 128);
    const int bn = I_PROC_NODE + 8 * i;
    pn1[i] = repack(bn + 0, 256, 128, 256, 128);
    pn2[i] = repack(bn + 1, 128, 128, 128, 128);
    pn3[i] = repack(bn + 2, 128, 128, 128, 128);
  }
  const _Float16* dc_w1 = repack(I_DEC + 0, 128, 128, 128, 128);
  const _Float16* dc_w2 = repack(I_DEC + 1, 128, 128, 128, 128);
  const _Float16* dc_w3 = repack(I_DEC + 2, 128, 3, 128, 16);

  const int nodeBlocks = (nNodes + TM - 1) / TM;   // 157
  const int edgeBlocks = (nEdges + TM - 1) / TM;   // 2500

  auto smMLP = [](int kp1, bool withIdx) -> size_t {
    return (size_t)TM * kp1 * 2                // bufIn (f16)
         + 2 * (size_t)TM * LATENT * 2         // actA/actB (f16)
         + (size_t)TM * LATENT * 4             // outF (f32)
         + 2 * (size_t)TM * 4                  // mu, rs
         + (withIdx ? 2 * (size_t)TM * 4 : 0); // sIdx, rIdx
  };

  // ---- encoder ----
  k_enc_node<<<nodeBlocks, NTHREADS, smMLP(64, false), stream>>>(
      node_feat, ptype, F(I_EMB),
      en_w1, F(I_ENC_NODE + 3), en_w2, F(I_ENC_NODE + 4), en_w3, F(I_ENC_NODE + 5),
      F(I_ENC_NODE + 6), F(I_ENC_NODE + 7), nodes, nNodes);

  k_enc_edge<<<edgeBlocks, NTHREADS, smMLP(32, false), stream>>>(
      edge_feat,
      ee_w1, F(I_ENC_EDGE + 3), ee_w2, F(I_ENC_EDGE + 4), ee_w3, F(I_ENC_EDGE + 5),
      F(I_ENC_EDGE + 6), F(I_ENC_EDGE + 7), edges, nEdges);

  // ---- processor: 10 message-passing steps ----
  const long long aggElems = (long long)nNodes * LATENT;
  for (int i = 0; i < 10; ++i) {
    const int be = I_PROC_EDGE + 8 * i;
    const int bn = I_PROC_NODE + 8 * i;

    k_zero<<<dim3((unsigned)((aggElems + 255) / 256)), dim3(256), 0, stream>>>(
        agg, aggElems);

    k_edge_step<<<edgeBlocks, NTHREADS, smMLP(384, true), stream>>>(
        nodes, edges, agg, senders, receivers,
        pe1[i], F(be + 3), pe2[i], F(be + 4), pe3[i], F(be + 5),
        F(be + 6), F(be + 7), nEdges);

    k_node_step<<<nodeBlocks, NTHREADS, smMLP(256, false), stream>>>(
        nodes, agg,
        pn1[i], F(bn + 3), pn2[i], F(bn + 4), pn3[i], F(bn + 5),
        F(bn + 6), F(bn + 7), nNodes);
  }

  // ---- decoder ----
  k_decode<<<nodeBlocks, NTHREADS, 3 * (size_t)TM * LATENT * 2, stream>>>(
      nodes, dc_w1, F(I_DEC + 3), dc_w2, F(I_DEC + 4), dc_w3, F(I_DEC + 5),
      (float*)d_out, nNodes);
}
